// MyModel_61933428413736
// MI455X (gfx1250) — matrix operand order repaired and checked by reading, not compiled, not run on hardware
//
#include <hip/hip_runtime.h>
#include <hip/hip_bf16.h>

// ---------------------------------------------------------------------------
// Spectral-norm LSTM, fused for MI455X (gfx1250, wave32, WMMA bf16->f32).
//   S=1024, B=64, I=H=512, 4H=2048.
// Pipeline (all on `stream`):
//   1. sigma_kernel x2   : one power iteration -> 1/sigma per weight matrix
//   2. pack_init_kernel  : W/sigma -> bf16, lane-contiguous WMMA B-fragment
//                          layout; combined bias; zero h buffers / barrier
//   3. xswz_kernel       : (if ws allows) x fp32 -> bf16, pre-swizzled into
//                          the per-chunk LDS staging image (one-off, parallel)
//   4. lstm_scan_kernel  : persistent 16-WG kernel, 1024 steps; fused
//                          x_t@Wih^T + h@Whh^T via v_wmma_f32_16x16x32_bf16;
//                          stage-ahead global_load_async_to_lds_b128 staging.
// ---------------------------------------------------------------------------

typedef __attribute__((ext_vector_type(16))) __bf16 v16bf;
typedef __attribute__((ext_vector_type(8)))  float  v8f;

#define S_LEN 1024
#define BATCH 64
#define HDIM  512
#define GDIM  2048   // 4*H
#define NWG   16     // persistent workgroups in the scan
#define CHUNK 2048   // elements per (64-row x 32-K) staged A chunk

__device__ __forceinline__ unsigned short f2bfu(float f) {
  unsigned u = __builtin_bit_cast(unsigned, f);
  unsigned r = u + 0x7FFFu + ((u >> 16) & 1u);   // round-to-nearest-even
  return (unsigned short)(r >> 16);
}

// direct async global -> LDS copy (16B per lane), tracked by ASYNCcnt
__device__ __forceinline__ void async_cp16(unsigned ldsOff, const void* g) {
  asm volatile("global_load_async_to_lds_b128 %0, %1, off"
               :: "v"(ldsOff), "v"(g) : "memory");
}
__device__ __forceinline__ void async_wait0() {
  asm volatile("s_wait_asynccnt 0x0" ::: "memory");
}

// ---------------------------------------------------------------------------
// One torch-style power iteration: v = l2n(W^T u); sigma = ||Wv||^2/(||Wv||+eps)
// (since u_new = Wv/(||Wv||+eps) and sigma = u_new . Wv). One WG, one-off cost.
// ---------------------------------------------------------------------------
__global__ void sigma_kernel(const float* __restrict__ W,
                             const float* __restrict__ u,
                             float* __restrict__ inv_sigma) {
  __shared__ float vS[HDIM];
  __shared__ float red[256];
  __shared__ float sc;
  const int tid = threadIdx.x;

  for (int j = tid; j < HDIM; j += 256) {            // v_raw = W^T u
    float s = 0.f;
    for (int g = 0; g < GDIM; ++g) s += W[(size_t)g * HDIM + j] * u[g];
    vS[j] = s;
  }
  __syncthreads();
  float loc = 0.f;
  for (int j = tid; j < HDIM; j += 256) loc += vS[j] * vS[j];
  red[tid] = loc; __syncthreads();
  for (int s = 128; s > 0; s >>= 1) { if (tid < s) red[tid] += red[tid + s]; __syncthreads(); }
  if (tid == 0) sc = 1.f / (sqrtf(red[0]) + 1e-12f);
  __syncthreads();
  const float scale = sc;
  for (int j = tid; j < HDIM; j += 256) vS[j] *= scale;
  __syncthreads();

  float s2 = 0.f;                                    // ||W v||^2
  for (int g = tid; g < GDIM; g += 256) {
    float d = 0.f;
    for (int j = 0; j < HDIM; ++j) d += W[(size_t)g * HDIM + j] * vS[j];
    s2 += d * d;
  }
  red[tid] = s2; __syncthreads();
  for (int s = 128; s > 0; s >>= 1) { if (tid < s) red[tid] += red[tid + s]; __syncthreads(); }
  if (tid == 0) {
    float nn    = sqrtf(red[0]);
    float sigma = red[0] / (nn + 1e-12f);
    *inv_sigma  = 1.f / sigma;
  }
}

// ---------------------------------------------------------------------------
// Pack W/sigma as bf16, lane-contiguous WMMA B-fragment layout:
//   fragment (kt, ntG) = 32x16 tile of W^T; lane L's 16 elements (K =
//   kt*32 + (e<8 ? 8h+e : 16+8h+e-8), h=L>=16, N = ntG*16 + (L&15)) -> one contiguous 32B blk.
// Also: bias = b_ih + b_hh, zero h double-buffer, zero barrier state.
// ---------------------------------------------------------------------------
__global__ void pack_init_kernel(const float* __restrict__ w_ih,
                                 const float* __restrict__ w_hh,
                                 const float* __restrict__ b_ih,
                                 const float* __restrict__ b_hh,
                                 const float* __restrict__ sig,
                                 unsigned short* __restrict__ wihP,
                                 unsigned short* __restrict__ whhP,
                                 float* __restrict__ bias,
                                 unsigned short* __restrict__ hG,
                                 int* __restrict__ bar) {
  const size_t NM = (size_t)GDIM * HDIM;
  size_t i = (size_t)blockIdx.x * 256 + threadIdx.x;
  if (i < 2 * NM) {
    const int    sel = (i >= NM);
    const size_t idx = sel ? (i - NM) : i;
    const int n = (int)(idx >> 9);          // row of W   (gate dim, 2048)
    const int k = (int)(idx & 511);         // col of W   (K dim, 512)
    const float v = (sel ? w_hh : w_ih)[idx] * sig[sel];
    const int kt  = k >> 5, kk = k & 31;
    const int ntG = n >> 4, cn = n & 15;
    const int lane = ((kk >> 3) & 1) * 16 + cn;
    const int e    = (kk & 7) + ((kk >> 4) << 3);
    const size_t off = ((size_t)(kt * 128 + ntG)) * 512 + lane * 16 + e;
    (sel ? whhP : wihP)[off] = f2bfu(v);
  }
  if (i < GDIM)                bias[i] = b_ih[i] + b_hh[i];
  if (i < 2 * BATCH * HDIM)    hG[i]   = 0;      // bf16 zero
  if (i < 64)                  bar[i]  = 0;      // barrier cnt/gen
}

// ---------------------------------------------------------------------------
// Pre-swizzle x into per-(t,kt) 4KB chunk images matching the LDS A-fragment
// layout: within a row's 32 K, 8-blocks stored {K0-7,K16-23,K8-15,K24-31}.
// Chunk (t,kt) element (m, perm(kk)) at xswz[t*32768 + kt*2048 + m*32 + ...].
// One 16B block per thread.
// ---------------------------------------------------------------------------
__global__ void xswz_kernel(const float* __restrict__ x,
                            unsigned short* __restrict__ xswz) {
  const unsigned j  = blockIdx.x * 256 + threadIdx.x;   // 16B block id
  const int cb  = j & 255;            // block within chunk
  const int kt  = (j >> 8) & 15;
  const int t   = j >> 12;
  const int m   = cb >> 2;
  const int pp  = (cb & 3) * 8;       // permuted position {0,8,16,24}
  const int skk = ((pp & 8) << 1) | ((pp & 16) >> 1);   // inverse perm
  const float* xr = x + (size_t)t * BATCH * HDIM + m * HDIM + kt * 32 + skk;
  float4 f0 = *(const float4*)xr;
  float4 f1 = *(const float4*)(xr + 4);
  uint4 pk;
  pk.x = f2bfu(f0.x) | ((unsigned)f2bfu(f0.y) << 16);
  pk.y = f2bfu(f0.z) | ((unsigned)f2bfu(f0.w) << 16);
  pk.z = f2bfu(f1.x) | ((unsigned)f2bfu(f1.y) << 16);
  pk.w = f2bfu(f1.z) | ((unsigned)f2bfu(f1.w) << 16);
  *(uint4*)&xswz[(size_t)t * 32768 + kt * CHUNK + cb * 8] = pk;
}

// ---------------------------------------------------------------------------
// Persistent recurrence. WG w owns h-columns [w*32, w*32+32) and the matching
// four gate-column blocks {g*512 + w*32 .. +31}. 8 waves: wave = N-tile (16
// gate cols), each wave does 4 M-tiles (M=64 batch).
// h is stored pre-permuted per chunk, so staging is an identity async copy.
// Stage-ahead pipeline: issue chunk kt+1 right after the barrier, then run
// chunk kt's 4 WMMAs while the async DMA fills the other LDS buffer.
// ---------------------------------------------------------------------------
template <bool PRE>
__global__ void __launch_bounds__(256, 1)
lstm_scan_kernel(const float* __restrict__ x,
                 const unsigned short* __restrict__ xswz,
                 const unsigned short* __restrict__ wihP,
                 const unsigned short* __restrict__ whhP,
                 const float* __restrict__ bias,
                 unsigned short* __restrict__ hG,
                 float* __restrict__ out,
                 int* __restrict__ bar) {
  __shared__ unsigned short As[2][CHUNK];        // 2 x 4KB A staging
  __shared__ float Gs[BATCH][128];               // gate pre-activations (32KB)
  __shared__ float cS[BATCH * 32];               // persistent cell state (8KB)

  const int tid  = threadIdx.x;
  const int wave = tid >> 5, lane = tid & 31;
  const int wg   = blockIdx.x;
  const int j0   = wg * 32;

  for (int i = tid; i < BATCH * 32; i += 256) cS[i] = 0.f;

  const int gsel  = wave >> 1;                             // gate (i,f,g,o)
  const int nGlob = gsel * HDIM + j0 + (wave & 1) * 16 + (lane & 15);
  const int ntG   = nGlob >> 4;                            // global N-tile id
  const float bc  = bias[nGlob];

  // staging (fallback path): thread -> (row sm, K block skk), permuted pos
  const int sm   = tid >> 2;
  const int skk  = (tid & 3) * 8;
  const int spos = ((skk & 8) << 1) | ((skk & 16) >> 1);
  // fragment read: lane's contiguous 16 elements within its row
  const int fm   = lane & 15;
  const int fseg = (lane >> 4) * 16;
  // identity-copy LDS offsets for async staging
  const unsigned ldsA0 = (unsigned)(size_t)&As[0][tid * 8];
  const unsigned ldsA1 = (unsigned)(size_t)&As[1][tid * 8];

  int* cnt = bar;
  int* gen = bar + 32;

  for (int t = 0; t < S_LEN; ++t) {
    const unsigned short* xc = xswz + (size_t)t * 32768;               // PRE
    const float*          xt = x + (size_t)t * BATCH * HDIM;           // !PRE
    const unsigned short* hP = hG + (size_t)( t      & 1) * BATCH * HDIM;
    unsigned short*       hN = hG + (size_t)((t + 1) & 1) * BATCH * HDIM;

    if (t + 1 < S_LEN) {                       // pull next x toward L2
      if (PRE) {
        const unsigned short* xn = xswz + (size_t)(t + 1) * 32768;
        for (int off = tid * 128; off < 32768; off += 256 * 128)
          __builtin_prefetch(xn + off, 0, 1);
      } else {
        const float* xn = x + (size_t)(t + 1) * BATCH * HDIM;
        for (int off = tid * 64; off < BATCH * HDIM; off += 256 * 64)
          __builtin_prefetch(xn + off, 0, 1);
      }
    }

    v8f acc[4] = {};

    // ---- phase 0: x_t contribution ----
    if (PRE) {
      async_cp16(ldsA0, xc + tid * 8);                 // prologue: chunk 0
#pragma unroll 4
      for (int kt = 0; kt < 16; ++kt) {
        async_wait0();
        __syncthreads();
        if (kt < 15)                                    // stage ahead
          async_cp16((kt & 1) ? ldsA0 : ldsA1, xc + (kt + 1) * CHUNK + tid * 8);
        const v16bf bfrag =
            *(const v16bf*)(wihP + ((size_t)(kt * 128 + ntG)) * 512 + lane * 16);
#pragma unroll
        for (int mt = 0; mt < 4; ++mt) {
          const v16bf afrag = *(const v16bf*)&As[kt & 1][(mt * 16 + fm) * 32 + fseg];
          acc[mt] = __builtin_amdgcn_wmma_f32_16x16x32_bf16(
              false, afrag, false, bfrag, (short)0, acc[mt], false, false);
        }
      }
    } else {
#pragma unroll 4
      for (int kt = 0; kt < 16; ++kt) {
        const float* xr = xt + sm * HDIM + kt * 32 + skk;
        float4 f0 = *(const float4*)xr;
        float4 f1 = *(const float4*)(xr + 4);
        uint4 pk;
        pk.x = f2bfu(f0.x) | ((unsigned)f2bfu(f0.y) << 16);
        pk.y = f2bfu(f0.z) | ((unsigned)f2bfu(f0.w) << 16);
        pk.z = f2bfu(f1.x) | ((unsigned)f2bfu(f1.y) << 16);
        pk.w = f2bfu(f1.z) | ((unsigned)f2bfu(f1.w) << 16);
        *(uint4*)&As[kt & 1][sm * 32 + spos] = pk;
        __syncthreads();
        const v16bf bfrag =
            *(const v16bf*)(wihP + ((size_t)(kt * 128 + ntG)) * 512 + lane * 16);
#pragma unroll
        for (int mt = 0; mt < 4; ++mt) {
          const v16bf afrag = *(const v16bf*)&As[kt & 1][(mt * 16 + fm) * 32 + fseg];
          acc[mt] = __builtin_amdgcn_wmma_f32_16x16x32_bf16(
              false, afrag, false, bfrag, (short)0, acc[mt], false, false);
        }
      }
    }

    // ---- phase 1: h_{t-1} contribution (h stored pre-permuted per chunk) ----
    async_cp16(ldsA0, hP + tid * 8);                   // prologue: chunk 0
#pragma unroll 4
    for (int kt = 0; kt < 16; ++kt) {
      async_wait0();
      __syncthreads();
      if (kt < 15)                                      // stage ahead
        async_cp16((kt & 1) ? ldsA0 : ldsA1, hP + (kt + 1) * CHUNK + tid * 8);
      const v16bf bfrag =
          *(const v16bf*)(whhP + ((size_t)(kt * 128 + ntG)) * 512 + lane * 16);
#pragma unroll
      for (int mt = 0; mt < 4; ++mt) {
        const v16bf afrag = *(const v16bf*)&As[kt & 1][(mt * 16 + fm) * 32 + fseg];
        acc[mt] = __builtin_amdgcn_wmma_f32_16x16x32_bf16(
            false, afrag, false, bfrag, (short)0, acc[mt], false, false);
      }
    }

    // scatter C tiles (+bias) into LDS gate buffer: cols [0,128) = i|f|g|o x32
    const int gcol = gsel * 32 + (wave & 1) * 16 + (lane & 15);
    const int rofs = (lane >> 4) * 8;
#pragma unroll
    for (int mt = 0; mt < 4; ++mt)
#pragma unroll
      for (int r = 0; r < 8; ++r)
        Gs[mt * 16 + r + rofs][gcol] = acc[mt][r] + bc;
    __syncthreads();

    for (int i = tid; i < BATCH * 32; i += 256) {
      const int b = i >> 5, jj = i & 31;
      float ig = 1.f / (1.f + __expf(-Gs[b][jj]));
      float fg = 1.f / (1.f + __expf(-Gs[b][32 + jj]));
      float gg = tanhf(Gs[b][64 + jj]);
      float og = 1.f / (1.f + __expf(-Gs[b][96 + jj]));
      float c  = fg * cS[i] + ig * gg;
      cS[i] = c;
      float h = og * tanhf(c);
      out[((size_t)t * BATCH + b) * HDIM + j0 + jj] = h;
      // pre-permuted h chunk image: chunk kt == wg for this WG's columns
      const int pj = ((jj & 8) << 1) | ((jj & 16) >> 1) | (jj & 7);
      hN[wg * CHUNK + b * 32 + pj] = f2bfu(h);
    }

    __threadfence();                     // publish hN before barrier release
    __syncthreads();
    if (tid == 0) {                      // device barrier across 16 WGs
      int g = __hip_atomic_load(gen, __ATOMIC_ACQUIRE, __HIP_MEMORY_SCOPE_AGENT);
      int a = __hip_atomic_fetch_add(cnt, 1, __ATOMIC_ACQ_REL, __HIP_MEMORY_SCOPE_AGENT);
      if (a == NWG - 1) {
        __hip_atomic_store(cnt, 0, __ATOMIC_RELAXED, __HIP_MEMORY_SCOPE_AGENT);
        __hip_atomic_fetch_add(gen, 1, __ATOMIC_RELEASE, __HIP_MEMORY_SCOPE_AGENT);
      } else {
        while (__hip_atomic_load(gen, __ATOMIC_ACQUIRE, __HIP_MEMORY_SCOPE_AGENT) == g)
          __builtin_amdgcn_s_sleep(1);
      }
    }
    __syncthreads();
  }
}

// ---------------------------------------------------------------------------
extern "C" void kernel_launch(void* const* d_in, const int* in_sizes, int n_in,
                              void* d_out, int out_size, void* d_ws, size_t ws_size,
                              hipStream_t stream) {
  const float* x    = (const float*)d_in[0];
  const float* w_ih = (const float*)d_in[1];
  const float* w_hh = (const float*)d_in[2];
  const float* b_ih = (const float*)d_in[3];
  const float* b_hh = (const float*)d_in[4];
  const float* u_ih = (const float*)d_in[5];
  const float* u_hh = (const float*)d_in[6];

  char* ws = (char*)d_ws;
  unsigned short* wihP = (unsigned short*)ws;                        // 2 MB
  unsigned short* whhP = (unsigned short*)(ws + (2u << 20));         // 2 MB
  float*          bias = (float*)(ws + (4u << 20));                  // 8 KB
  float*          sig  = (float*)(ws + (4u << 20) + 8192);           // 2 floats
  int*            bar  = (int*)  (ws + (4u << 20) + 8448);           // 64 ints
  unsigned short* hG   = (unsigned short*)(ws + (4u << 20) + 16384); // 128 KB
  unsigned short* xswz = (unsigned short*)(ws + (8u << 20));         // 64 MB

  sigma_kernel<<<1, 256, 0, stream>>>(w_ih, u_ih, sig + 0);
  sigma_kernel<<<1, 256, 0, stream>>>(w_hh, u_hh, sig + 1);

  const int packBlocks = (int)((2ull * GDIM * HDIM + 255) / 256);
  pack_init_kernel<<<packBlocks, 256, 0, stream>>>(w_ih, w_hh, b_ih, b_hh, sig,
                                                   wihP, whhP, bias, hG, bar);

  const size_t needPre = (8ull << 20) + ((size_t)S_LEN * 32768 * 2);
  if (ws_size >= needPre) {
    xswz_kernel<<<(S_LEN * 16 * 256) / 256, 256, 0, stream>>>(x, xswz);
    lstm_scan_kernel<true><<<NWG, 256, 0, stream>>>(x, xswz, wihP, whhP, bias,
                                                    hG, (float*)d_out, bar);
  } else {
    lstm_scan_kernel<false><<<NWG, 256, 0, stream>>>(x, xswz, wihP, whhP, bias,
                                                     hG, (float*)d_out, bar);
  }
}
